// MultiheadAttention_80307298501404
// MI455X (gfx1250) — matrix operand order repaired and checked by reading, not compiled, not run on hardware
//
#include <hip/hip_runtime.h>

// ---------- types ----------
typedef __attribute__((ext_vector_type(16))) __bf16          v16bf;
typedef __attribute__((ext_vector_type(8)))  float           v8f;
typedef __attribute__((ext_vector_type(8)))  unsigned short  v8u;
typedef __attribute__((ext_vector_type(4))) unsigned int     tdm_v4u;
typedef __attribute__((ext_vector_type(8))) int              tdm_v8i;
typedef __attribute__((ext_vector_type(4))) int              tdm_v4i;

union frag {
    v16bf bf;
    v8u   u8[2];
    unsigned short us[16];
};

__device__ __forceinline__ unsigned short f2bf(float f) {
    unsigned int u = __builtin_bit_cast(unsigned int, f);
    u += 0x7FFFu + ((u >> 16) & 1u);          // round-to-nearest-even
    return (unsigned short)(u >> 16);
}
__device__ __forceinline__ float bf2f(unsigned short h) {
    unsigned int u = ((unsigned int)h) << 16;
    return __builtin_bit_cast(float, u);
}

// ---------- Tensor Data Mover (gfx1250), availability per compile pass ----------
#if defined(__AMDGCN__) && defined(__has_builtin)
#if __has_builtin(__builtin_amdgcn_tensor_load_to_lds) && \
    __has_builtin(__builtin_amdgcn_s_wait_tensorcnt)
#define HAVE_TDM 1
#endif
#endif
#ifndef HAVE_TDM
#define HAVE_TDM 0
#endif

// Always DECLARED (so host-pass template parsing resolves the name);
// the builtin call only exists in passes where it is available.
// 2D tile DMA: global (row-major, stride0 elems of 2B) -> LDS, with LDS row
// padding of 4 DWORDs after every 16 DWORDs (matches our 32+8 ushort rows).
__device__ __forceinline__ void tdm_load_tile_2d(
    const unsigned short* gtile, unsigned lds_addr,
    unsigned tensor_d0, unsigned tensor_d1,
    unsigned long long stride0, unsigned tile_d0, unsigned tile_d1)
{
#if HAVE_TDM
    unsigned long long ga = (unsigned long long)(const void*)gtile;
    tdm_v4u g0;
    g0[0] = 1u;                                   // count=1, user mode
    g0[1] = lds_addr;                             // LDS byte address
    g0[2] = (unsigned)ga;                         // global_addr[31:0]
    g0[3] = (unsigned)((ga >> 32) & 0x01FFFFFFu)  // global_addr[56:32]
            | (2u << 30);                         // type = 2 ("image")
    unsigned w[8] = {};
    w[0] = (1u << 16)        // data_size = 2 bytes
         | (1u << 20)        // pad_enable
         | (3u << 22)        // pad_interval: 16 DWORDs (64B = one 32xbf16 row)
         | (3u << 25);       // pad_amount: 4 DWORDs (16B = 8 ushort pad)
    w[1] = (tensor_d0 & 0xFFFFu) << 16;                  // tensor_dim0[15:0]
    w[2] = ((tensor_d0 >> 16) & 0xFFFFu)                 // tensor_dim0[31:16]
         | ((tensor_d1 & 0xFFFFu) << 16);                // tensor_dim1[15:0]
    w[3] = ((tensor_d1 >> 16) & 0xFFFFu)                 // tensor_dim1[31:16]
         | ((tile_d0 & 0xFFFFu) << 16);                  // tile_dim0
    w[4] = (tile_d1 & 0xFFFFu);                          // tile_dim1 (tile_dim2=0)
    w[5] = (unsigned)(stride0 & 0xFFFFFFFFull);          // tensor_dim0_stride[31:0]
    w[6] = (unsigned)((stride0 >> 32) & 0xFFFFull);      // tensor_dim0_stride[47:32]
    tdm_v8i g1;
    #pragma unroll
    for (int i = 0; i < 8; ++i) g1[i] = (int)w[i];
    tdm_v4i z4 = {0, 0, 0, 0};
#if __clang_major__ >= 23
    tdm_v8i z8 = {0, 0, 0, 0, 0, 0, 0, 0};
    __builtin_amdgcn_tensor_load_to_lds(g0, g1, z4, z4, z8, 0);
#else
    __builtin_amdgcn_tensor_load_to_lds(g0, g1, z4, z4, 0);
#endif
#else
    (void)gtile; (void)lds_addr; (void)tensor_d0; (void)tensor_d1;
    (void)stride0; (void)tile_d0; (void)tile_d1;
#endif
}

__device__ __forceinline__ void tdm_wait_all()
{
#if HAVE_TDM
    __builtin_amdgcn_s_wait_tensorcnt(0);
#endif
}

// ---------------------------------------------------------------------------
// Tiled bf16 WMMA GEMM. Block tile 128(M) x 256(N), 256 threads = 8 waves
// (wave grid 4(M) x 2(N)); wave tile 32x128 = 2x8 WMMA accumulators.
// K-step 32, double-buffered LDS; TDM staging for straight bf16 tile copies.
// MODE 0: A fp32, B fp32 (weights), +bias, C bf16          (input projection)
// MODE 1: A bf16, B(k,n) = Kmat[n*K + k] (QK^T), C = bf16(acc*scale)  [TDM A+B]
// MODE 2: A bf16, B bf16 row-major, C bf16 COLUMN-major (fused transpose) [TDM A]
// MODE 3: A bf16, B fp32 (weights), +bias, C fp32          (output proj) [TDM A]
// M % 128 == 0, N % 256 == 0, K % 32 == 0 for all launches below.
// ---------------------------------------------------------------------------
template <int MODE>
__global__ __launch_bounds__(256) void gemm_wmma(
    const void* __restrict__ Ap, const void* __restrict__ Bp,
    const float* __restrict__ bias, void* __restrict__ Cp,
    int M, int N, int K,
    long long strideA, long long strideB, long long strideC, float scale)
{
    constexpr int AS = 40;                 // LDS row stride: 32 data + 8 pad
    constexpr int BS = 40;
    __shared__ __align__(16) unsigned short shA[2][128 * AS];  // [m][k] bf16
    __shared__ __align__(16) unsigned short shB[2][256 * BS];  // [n][k] bf16

    const int t    = threadIdx.x;
    const int lane = t & 31;
    const int wave = t >> 5;
    const int wm   = wave & 3;             // 0..3 -> 32 rows each
    const int wn   = wave >> 2;            // 0..1 -> 128 cols each
    const int z    = blockIdx.z;
    const int half = lane >> 4;
    const int l16  = lane & 15;

    const long long gm0 = (long long)blockIdx.y * 128;
    const long long gn0 = (long long)blockIdx.x * 256;

    const float*          Af = (const float*)Ap          + (long long)z * strideA;
    const unsigned short* Ab = (const unsigned short*)Ap + (long long)z * strideA;
    const float*          Bf = (const float*)Bp          + (long long)z * strideB;
    const unsigned short* Bb = (const unsigned short*)Bp + (long long)z * strideB;
    float*                Cf = (float*)Cp                + (long long)z * strideC;
    unsigned short*       Cb = (unsigned short*)Cp       + (long long)z * strideC;

    constexpr bool TDM_A = HAVE_TDM && (MODE == 1 || MODE == 2 || MODE == 3);
    constexpr bool TDM_B = HAVE_TDM && (MODE == 1);

    v8f acc[2][8] = {};

    // staging registers (load phase -> store phase, overlapping WMMA compute)
    float          aF[16]; unsigned short aH[16];
    float          bF[32]; unsigned short bH[32];

    // ---- load phase: global -> regs (or TDM issue), + prefetch k+64 ----
    auto load_phase = [&](int kk, int nbuf) {
        // A tile: 128 x 32
        if constexpr (MODE == 0) {
            #pragma unroll
            for (int i = 0; i < 16; ++i) {
                int idx = t + i * 256, r = idx >> 5, c = idx & 31;
                aF[i] = Af[(gm0 + r) * (long long)K + kk + c];
            }
            if (kk + 32 < K)
                __builtin_prefetch(&Af[(gm0 + (t & 127)) * (long long)K +
                                       kk + 32 + (t >> 7) * 16], 0, 3);
        } else if constexpr (TDM_A) {
            if (wave == 0)
                tdm_load_tile_2d(&Ab[gm0 * (long long)K + kk],
                                 (unsigned)(size_t)&shA[nbuf][0],
                                 (unsigned)K, (unsigned)M,
                                 (unsigned long long)K, 32u, 128u);
        } else {
            #pragma unroll
            for (int i = 0; i < 16; ++i) {
                int idx = t + i * 256, r = idx >> 5, c = idx & 31;
                aH[i] = Ab[(gm0 + r) * (long long)K + kk + c];
            }
        }
        // B tile, staged as [n][k]: 256 x 32
        if constexpr (MODE == 0 || MODE == 3) {      // fp32 weights, transpose
            #pragma unroll
            for (int i = 0; i < 32; ++i)
                bF[i] = Bf[(long long)(kk + i) * N + gn0 + t];
            if (kk + 32 < K)
                __builtin_prefetch(&Bf[(long long)(kk + 32 + (t & 31)) * N +
                                       gn0 + (t >> 5) * 32], 0, 3);
        } else if constexpr (TDM_B) {                // QK^T: straight tile copy
            if (wave == 1)
                tdm_load_tile_2d(&Bb[gn0 * (long long)K + kk],
                                 (unsigned)(size_t)&shB[nbuf][0],
                                 (unsigned)K, (unsigned)N,
                                 (unsigned long long)K, 32u, 256u);
        } else if constexpr (MODE == 1) {            // manual fallback
            #pragma unroll
            for (int i = 0; i < 32; ++i) {
                int idx = t + i * 256, n = idx >> 5, c = idx & 31;
                bH[i] = Bb[(gn0 + n) * (long long)K + kk + c];
            }
        } else {                                     // MODE 2: bf16 transpose
            #pragma unroll
            for (int i = 0; i < 32; ++i)
                bH[i] = Bb[(long long)(kk + i) * N + gn0 + t];
            if (kk + 32 < K)
                __builtin_prefetch(&Bb[(long long)(kk + 32 + (t & 31)) * N +
                                       gn0 + (t >> 5) * 32], 0, 3);
        }
    };

    // ---- store phase: regs -> LDS (manual paths only) ----
    auto store_phase = [&](int nbuf) {
        if constexpr (MODE == 0) {
            #pragma unroll
            for (int i = 0; i < 16; ++i) {
                int idx = t + i * 256, r = idx >> 5, c = idx & 31;
                shA[nbuf][r * AS + c] = f2bf(aF[i]);
            }
        } else if constexpr (!TDM_A) {
            #pragma unroll
            for (int i = 0; i < 16; ++i) {
                int idx = t + i * 256, r = idx >> 5, c = idx & 31;
                shA[nbuf][r * AS + c] = aH[i];
            }
        }
        if constexpr (MODE == 0 || MODE == 3) {
            #pragma unroll
            for (int i = 0; i < 32; ++i)
                shB[nbuf][t * BS + i] = f2bf(bF[i]);
        } else if constexpr (MODE == 1 && !TDM_B) {
            #pragma unroll
            for (int i = 0; i < 32; ++i) {
                int idx = t + i * 256, n = idx >> 5, c = idx & 31;
                shB[nbuf][n * BS + c] = bH[i];
            }
        } else if constexpr (MODE == 2) {
            #pragma unroll
            for (int i = 0; i < 32; ++i)
                shB[nbuf][t * BS + i] = bH[i];
        }
    };

    auto tile_barrier = [&]() {
        if constexpr (TDM_A || TDM_B) tdm_wait_all();
        __syncthreads();
    };

    // ---- compute: 2 A frags, 8 streamed B frags, 16 WMMAs per K-step ----
    auto compute = [&](int bsel) {
        frag fa[2];
        #pragma unroll
        for (int im = 0; im < 2; ++im) {
            int m = wm * 32 + im * 16 + l16;
            fa[im].u8[0] = *(const v8u*)&shA[bsel][m * AS + half * 8];
            fa[im].u8[1] = *(const v8u*)&shA[bsel][m * AS + 16 + half * 8];
        }
        #pragma unroll
        for (int jn = 0; jn < 8; ++jn) {
            int n = wn * 128 + jn * 16 + l16;
            frag fb;
            fb.u8[0] = *(const v8u*)&shB[bsel][n * BS + half * 8];
            fb.u8[1] = *(const v8u*)&shB[bsel][n * BS + 16 + half * 8];
            #pragma unroll
            for (int im = 0; im < 2; ++im)
                acc[im][jn] = __builtin_amdgcn_wmma_f32_16x16x32_bf16(
                    false, fa[im].bf, false, fb.bf,
                    (short)0, acc[im][jn], false, false);
        }
    };

    // ---- main pipeline: double-buffered LDS ----
    load_phase(0, 0);
    store_phase(0);
    tile_barrier();

    int buf = 0;
    for (int k0 = 0; k0 < K; k0 += 32) {
        const bool has_next = (k0 + 32) < K;
        if (has_next) load_phase(k0 + 32, buf ^ 1);   // overlap with WMMAs
        compute(buf);
        if (has_next) store_phase(buf ^ 1);
        tile_barrier();
        buf ^= 1;
    }

    // ---- epilogue: C 16x16 f32 layout: VGPR r -> M=r (lanes0-15)/M=8+r ----
    #pragma unroll
    for (int im = 0; im < 2; ++im) {
        #pragma unroll
        for (int jn = 0; jn < 8; ++jn) {
            long long gmb = gm0 + wm * 32 + im * 16 + half * 8;
            long long gc  = gn0 + wn * 128 + jn * 16 + l16;
            float bv = 0.0f;
            if constexpr (MODE == 0 || MODE == 3) bv = bias[gc];
            #pragma unroll
            for (int r = 0; r < 8; ++r) {
                long long gm = gmb + r;
                float v = acc[im][jn][r];
                if constexpr (MODE == 0)
                    Cb[gm * (long long)N + gc] = f2bf(v + bv);
                else if constexpr (MODE == 1)
                    Cb[gm * (long long)N + gc] = f2bf(v * scale);
                else if constexpr (MODE == 2)
                    Cb[gc * (long long)M + gm] = f2bf(v);     // column-major
                else
                    Cf[gm * (long long)N + gc] = v + bv;
            }
        }
    }
}

// ---------------------------------------------------------------------------
// Row-wise softmax over bf16 scores, in place. One block per row of 2048.
// ---------------------------------------------------------------------------
__global__ __launch_bounds__(256) void softmax_rows(unsigned short* __restrict__ sc,
                                                    int ncol)
{
    __shared__ float red[256];
    const int t = threadIdx.x;
    const long long base = (long long)blockIdx.x * ncol;

    float x[8];
    float mx = -3.0e38f;
    #pragma unroll
    for (int i = 0; i < 8; ++i) {
        x[i] = bf2f(sc[base + t + i * 256]);
        mx = fmaxf(mx, x[i]);
    }
    red[t] = mx; __syncthreads();
    for (int s = 128; s > 0; s >>= 1) {
        if (t < s) red[t] = fmaxf(red[t], red[t + s]);
        __syncthreads();
    }
    mx = red[0]; __syncthreads();

    float sum = 0.0f;
    #pragma unroll
    for (int i = 0; i < 8; ++i) { x[i] = __expf(x[i] - mx); sum += x[i]; }
    red[t] = sum; __syncthreads();
    for (int s = 128; s > 0; s >>= 1) {
        if (t < s) red[t] += red[t + s];
        __syncthreads();
    }
    float inv = 1.0f / red[0];
    #pragma unroll
    for (int i = 0; i < 8; ++i)
        sc[base + t + i * 256] = f2bf(x[i] * inv);
}

// ---------------------------------------------------------------------------
extern "C" void kernel_launch(void* const* d_in, const int* in_sizes, int n_in,
                              void* d_out, int out_size, void* d_ws, size_t ws_size,
                              hipStream_t stream)
{
    (void)in_sizes; (void)n_in; (void)out_size; (void)ws_size;

    constexpr int B = 4, S = 2048, D = 1024;
    constexpr long long SD = (long long)S * D;   // 2 Mi elems
    constexpr long long SS = (long long)S * S;   // 4 Mi elems
    constexpr int M  = B * S;                    // 8192 fused rows

    const float* q  = (const float*)d_in[0];
    const float* k  = (const float*)d_in[1];
    const float* v  = (const float*)d_in[2];
    // d_in[3] = mask: discarded by the (buggy) reference
    const float* Wq = (const float*)d_in[4];
    const float* bq = (const float*)d_in[5];
    const float* Wk = (const float*)d_in[6];
    const float* bk = (const float*)d_in[7];
    const float* Wv = (const float*)d_in[8];
    const float* bv = (const float*)d_in[9];
    const float* Wo = (const float*)d_in[10];
    const float* bo = (const float*)d_in[11];

    // workspace layout (bf16 elems): Qb | Kb | Vb | Xp | Sc  (= 96 MB total)
    unsigned short* Qb = (unsigned short*)d_ws;
    unsigned short* Kb = Qb + (long long)M * D;
    unsigned short* Vb = Kb + (long long)M * D;
    unsigned short* Xp = Vb + (long long)M * D;   // attn@V, stored transposed
    unsigned short* Sc = Xp + (long long)M * D;   // scores / attn, [B][S][S]

    dim3 blk(256);
    const float qk_scale = 0.03125f;              // 1/sqrt(1024)

    // 1) input projections: [8192,1024] = fp32 @ fp32 + bias -> bf16
    gemm_wmma<0><<<dim3(4, 64, 1), blk, 0, stream>>>(q, Wq, bq, Qb, M, D, D, 0, 0, 0, 1.f);
    gemm_wmma<0><<<dim3(4, 64, 1), blk, 0, stream>>>(k, Wk, bk, Kb, M, D, D, 0, 0, 0, 1.f);
    gemm_wmma<0><<<dim3(4, 64, 1), blk, 0, stream>>>(v, Wv, bv, Vb, M, D, D, 0, 0, 0, 1.f);

    // 2) scores = (Q @ K^T) * scale, per batch: [2048,2048] bf16  (TDM staged)
    gemm_wmma<1><<<dim3(8, 16, B), blk, 0, stream>>>(Qb, Kb, nullptr, Sc,
                                                     S, S, D, SD, SD, SS, qk_scale);

    // 3) softmax over rows, in place
    softmax_rows<<<dim3(B * S), blk, 0, stream>>>(Sc, S);

    // 4) x = attn @ V, per batch, stored column-major (== reference transpose)
    gemm_wmma<2><<<dim3(4, 16, B), blk, 0, stream>>>(Sc, Vb, nullptr, Xp,
                                                     S, D, S, SS, SD, SD, 1.f);

    // 5) out = y @ Wo + bo -> fp32 d_out
    gemm_wmma<3><<<dim3(4, 64, 1), blk, 0, stream>>>(Xp, Wo, bo, (float*)d_out,
                                                     M, D, D, 0, 0, 0, 1.f);
}
